// ParsevalWaveletAttention_49641232007195
// MI455X (gfx1250) — compile-verified, matrix operand order checked by reading, not run on hardware
//
#include <hip/hip_runtime.h>

typedef unsigned short u16;
typedef __attribute__((ext_vector_type(16))) __bf16 v16bf;
typedef __attribute__((ext_vector_type(8)))  float  v8f;
typedef __attribute__((ext_vector_type(2)))  float  v2f;
typedef __attribute__((ext_vector_type(4)))  unsigned int v4u;

#define Bc 4
#define Tc 1024
#define Cc 768
#define Hc 12
#define Dc 64
#define HALFc 32
#define NEARc 64
#define NCc 15
#define NCP 16     // padded Haar column stride
#define Nh 48      // B*H
#define BTc 4096   // B*T

union V16 { v16bf v; u16 u[16]; v4u q[2]; };

__device__ __forceinline__ u16 f2bf(float f) {
  unsigned u = __float_as_uint(f);
  unsigned r = (u + 0x7FFFu + ((u >> 16) & 1u)) >> 16;
  return (u16)r;
}
__device__ __forceinline__ float bf2f(u16 x) {
  return __uint_as_float(((unsigned)x) << 16);
}
// CDNA5 16-bit A/B fragment K-mapping: lane-group g (lane>>4), half index h (0..15)
// K = (h&7) + 8*g + 16*(h>>3)  ->  per lane: 16B segments at [8g, +8) and [16+8g, +8)
__device__ __forceinline__ int kmap(int h, int g) {
  return (h & 7) + 8 * g + 16 * (h >> 3);
}
// Build a 16-half fragment from a row-major row with two 16-byte vector loads.
__device__ __forceinline__ void load_frag16(V16& f, const u16* rowbase, int g) {
  f.q[0] = *(const v4u*)(rowbase + 8 * g);
  f.q[1] = *(const v4u*)(rowbase + 16 + 8 * g);
}
__device__ __forceinline__ v8f wmma_bf16(v16bf a, v16bf b, v8f c) {
  return __builtin_amdgcn_wmma_f32_16x16x32_bf16(false, a, false, b, (short)0, c, false, false);
}

// ---------------------------------------------------------------- tables
__global__ __launch_bounds__(256) void build_tables(
    float* __restrict__ whaar, float* __restrict__ cosT, float* __restrict__ sinT,
    const float* __restrict__ k_null, const float* __restrict__ log_tau,
    float* __restrict__ kn, float* __restrict__ tau) {
  int idx = blockIdx.x * 256 + threadIdx.x;
  if (idx < Tc * NCP) {
    int t = idx / NCP, c = idx % NCP;
    float val = 0.f;
    if (c < NCc) {
      int j = 31 - __clz(c + 1);
      int k = (c + 1) - (1 << j);
      int bs = Tc >> j, halfb = bs >> 1, s = k * bs;
      float inv = 1.0f / sqrtf((float)halfb);
      float v = 0.f;
      if (t >= s && t < s + halfb) v = inv;
      else if (t >= s + halfb && t < s + bs) v = -inv;
      val = v / (sqrtf(2.0f) + 1e-6f);  // column norm is exactly sqrt(2)
    }
    whaar[idx] = val;
  }
  int i2 = idx - Tc * NCP;
  if (i2 >= 0 && i2 < Tc * HALFc) {
    int t = i2 / HALFc, i = i2 % HALFc;
    float invf = powf(10000.0f, -((float)i) / (float)HALFc);
    float ang = (float)t * invf;
    cosT[i2] = cosf(ang);
    sinT[i2] = sinf(ang);
  }
  int i3 = idx - Tc * NCP - Tc * HALFc;
  if (i3 >= 0 && i3 < Hc) {
    float lt = log_tau[i3];
    float sp = (lt > 20.f) ? lt : log1pf(expf(lt));
    tau[i3] = sp + 1e-4f;
    float s2 = 0.f;
    for (int d = 0; d < Dc; ++d) { float v = k_null[i3 * Dc + d]; s2 += v * v; }
    float inv = 1.0f / (sqrtf(s2) + 1e-6f);
    for (int d = 0; d < Dc; ++d) kn[i3 * Dc + d] = k_null[i3 * Dc + d] * inv;
  }
}

// ------------------------------------------------- layernorm + bf16 pack
__global__ __launch_bounds__(256) void pack_x_ln(
    const float* __restrict__ x, const float* __restrict__ ln_w, const float* __restrict__ ln_b,
    u16* __restrict__ x_bf, u16* __restrict__ xln_bf) {
  int row = blockIdx.x, tid = threadIdx.x;
  const float* xr = x + (size_t)row * Cc;
  __shared__ float red[256];
  float s = 0.f;
  for (int i = tid; i < Cc; i += 256) s += xr[i];
  red[tid] = s; __syncthreads();
  for (int o = 128; o > 0; o >>= 1) { if (tid < o) red[tid] += red[tid + o]; __syncthreads(); }
  float mean = red[0] / (float)Cc;
  __syncthreads();
  float s2 = 0.f;
  for (int i = tid; i < Cc; i += 256) { float d = xr[i] - mean; s2 += d * d; }
  red[tid] = s2; __syncthreads();
  for (int o = 128; o > 0; o >>= 1) { if (tid < o) red[tid] += red[tid + o]; __syncthreads(); }
  float inv = rsqrtf(red[0] / (float)Cc + 1e-5f);
  for (int i = tid; i < Cc; i += 256) {
    float xv = xr[i];
    x_bf[(size_t)row * Cc + i] = f2bf(xv);
    xln_bf[(size_t)row * Cc + i] = f2bf((xv - mean) * inv * ln_w[i] + ln_b[i]);
  }
}

// ------------------------------------ weight bf16 pack (no transpose needed:
// row n of W *is* column n of B = W^T for the fragment loader)
__global__ __launch_bounds__(256) void pack_wt(
    const float* __restrict__ Wq, const float* __restrict__ Wk,
    const float* __restrict__ Wv, const float* __restrict__ Wo,
    u16* __restrict__ Wqb, u16* __restrict__ Wkb,
    u16* __restrict__ Wvb, u16* __restrict__ Wob) {
  int idx = blockIdx.x * 256 + threadIdx.x;
  int mat = idx / (Cc * Cc), rem = idx % (Cc * Cc);
  if (mat >= 4) return;
  const float* src = (mat == 0) ? Wq : (mat == 1) ? Wk : (mat == 2) ? Wv : Wo;
  u16* dst = (mat == 0) ? Wqb : (mat == 1) ? Wkb : (mat == 2) ? Wvb : Wob;
  dst[rem] = f2bf(src[rem]);
}

// ---------------------------------------------------- bf16 WMMA GEMM
// C(MxN,f32) = A(MxK bf16 row-major) @ B, where Bw is [N][K] bf16 (B = Bw^T).
// 128 threads / 4 waves; each wave owns a 32x32 C tile (4 accumulators).
// Register double-buffering: fragments for k+32 are loaded into fresh regs
// before the WMMAs on the current fragments issue (partial loadcnt waits).
__global__ __launch_bounds__(128) void gemm_bf16_wmma(
    const u16* __restrict__ A, const u16* __restrict__ Bw, float* __restrict__ Cm,
    int M, int N, int K) {
  int tid = threadIdx.x;
  int lane = tid & 31, wave = tid >> 5;
  int g = lane >> 4, lm = lane & 15;
  int wm = wave >> 1, wn = wave & 1;
  int bm0 = blockIdx.y * 64 + wm * 32;
  int bn0 = blockIdx.x * 64 + wn * 32;
  const u16* arow0 = A  + (size_t)(bm0 + lm) * K;
  const u16* arow1 = A  + (size_t)(bm0 + 16 + lm) * K;
  const u16* brow0 = Bw + (size_t)(bn0 + lm) * K;
  const u16* brow1 = Bw + (size_t)(bn0 + 16 + lm) * K;
  const v8f vz = {0.f,0.f,0.f,0.f,0.f,0.f,0.f,0.f};
  v8f acc00 = vz, acc01 = vz, acc10 = vz, acc11 = vz;

  V16 a0c, a1c, b0c, b1c;
  load_frag16(a0c, arow0, g);
  load_frag16(a1c, arow1, g);
  load_frag16(b0c, brow0, g);
  load_frag16(b1c, brow1, g);
#pragma unroll 2
  for (int k0 = 32; k0 < K; k0 += 32) {
    if (k0 + 96 < K) {
      __builtin_prefetch((const char*)(arow0 + k0 + 96), 0, 1);
      __builtin_prefetch((const char*)(arow1 + k0 + 96), 0, 1);
      __builtin_prefetch((const char*)(brow0 + k0 + 96), 0, 1);
      __builtin_prefetch((const char*)(brow1 + k0 + 96), 0, 1);
    }
    V16 a0n, a1n, b0n, b1n;
    load_frag16(a0n, arow0 + k0, g);
    load_frag16(a1n, arow1 + k0, g);
    load_frag16(b0n, brow0 + k0, g);
    load_frag16(b1n, brow1 + k0, g);
    acc00 = wmma_bf16(a0c.v, b0c.v, acc00);
    acc01 = wmma_bf16(a0c.v, b1c.v, acc01);
    acc10 = wmma_bf16(a1c.v, b0c.v, acc10);
    acc11 = wmma_bf16(a1c.v, b1c.v, acc11);
    a0c = a0n; a1c = a1n; b0c = b0n; b1c = b1n;
  }
  acc00 = wmma_bf16(a0c.v, b0c.v, acc00);
  acc01 = wmma_bf16(a0c.v, b1c.v, acc01);
  acc10 = wmma_bf16(a1c.v, b0c.v, acc10);
  acc11 = wmma_bf16(a1c.v, b1c.v, acc11);

#pragma unroll
  for (int r = 0; r < 8; ++r) {
    int m0 = bm0 + r + 8 * g;
    int n0 = bn0 + lm;
    Cm[(size_t)m0 * N + n0]             = acc00[r];
    Cm[(size_t)m0 * N + n0 + 16]        = acc01[r];
    Cm[(size_t)(m0 + 16) * N + n0]      = acc10[r];
    Cm[(size_t)(m0 + 16) * N + n0 + 16] = acc11[r];
  }
}

// ------------------------------- rope + l2norm + vn + Sv, bf16 head layouts
__global__ __launch_bounds__(64) void post_qkv(
    const float* __restrict__ Qf, const float* __restrict__ Kf, const float* __restrict__ Vf,
    const float* __restrict__ x, const float* __restrict__ cosT, const float* __restrict__ sinT,
    const float* __restrict__ Amat,
    u16* __restrict__ qb, u16* __restrict__ kb, u16* __restrict__ vtb,
    u16* __restrict__ vnb, u16* __restrict__ svb) {
  int blk = blockIdx.x;
  int nh = blk / Tc, t = blk % Tc;
  int h = nh % Hc, b = nh / Hc;
  int d = threadIdx.x;
  size_t xoff = ((size_t)(b * Tc + t)) * Cc + h * Dc + d;
  float qv = Qf[xoff], kv = Kf[xoff], vv = Vf[xoff], xv = x[xoff];
  __shared__ float qs[64], ks[64], r0[64], r1[64], r2[64], vns[64];
  qs[d] = qv; ks[d] = kv;
  __syncthreads();
  float qr, kr;
  if (d < HALFc) {
    float c = cosT[t * HALFc + d], s = sinT[t * HALFc + d];
    qr = qs[d] * c - qs[d + HALFc] * s;
    kr = ks[d] * c - ks[d + HALFc] * s;
  } else {
    float c = cosT[t * HALFc + d - HALFc], s = sinT[t * HALFc + d - HALFc];
    qr = qs[d - HALFc] * s + qs[d] * c;
    kr = ks[d - HALFc] * s + ks[d] * c;
  }
  r0[d] = qr * qr; r1[d] = kr * kr; r2[d] = xv * xv;
  __syncthreads();
  for (int o = 32; o > 0; o >>= 1) {
    if (d < o) { r0[d] += r0[d + o]; r1[d] += r1[d + o]; r2[d] += r2[d + o]; }
    __syncthreads();
  }
  float qo = qr / (sqrtf(r0[0]) + 1e-6f);
  float ko = kr / (sqrtf(r1[0]) + 1e-6f);
  float vnv = xv * rsqrtf(r2[0] + 1e-6f);
  vns[d] = vnv;
  __syncthreads();
  const float* Ah = Amat + (size_t)h * Dc * Dc;
  float sv = 0.f;
  for (int dd = 0; dd < Dc; ++dd)
    sv += vns[dd] * (Ah[dd * Dc + d] - Ah[d * Dc + dd]);
  size_t o = ((size_t)nh * Tc + t) * Dc + d;
  qb[o] = f2bf(qo); kb[o] = f2bf(ko);
  vnb[o] = f2bf(vnv); svb[o] = f2bf(sv);
  vtb[((size_t)nh * Dc + d) * Tc + t] = f2bf(vv);   // V transposed: [nh][d][t]
}

// ----------------------------------------- Haar projection: comp[n][c][m]
__global__ __launch_bounds__(256) void haar_qkfar(
    const u16* __restrict__ qb, const u16* __restrict__ kb,
    const float* __restrict__ whaar, float* __restrict__ comp) {
  int n = blockIdx.x, tid = threadIdx.x;
  __shared__ float qfar[NCc][Dc], kfar[NCc][Dc];
  for (int i = tid; i < NCc * Dc; i += 256) {
    int c = i / Dc, d = i % Dc;
    float aq = 0.f, ak = 0.f;
    for (int t = 0; t < Tc; ++t) {
      float w = whaar[t * NCP + c];
      aq += w * bf2f(qb[((size_t)n * Tc + t) * Dc + d]);
      ak += w * bf2f(kb[((size_t)n * Tc + t) * Dc + d]);
    }
    qfar[c][d] = aq; kfar[c][d] = ak;
  }
  __syncthreads();
  for (int i = tid; i < NCc * NCc; i += 256) {
    int c = i / NCc, m = i % NCc;
    float a = 0.f;
    for (int d = 0; d < Dc; ++d) a += qfar[c][d] * kfar[m][d];
    comp[(size_t)n * NCc * NCc + i] = a;
  }
}

// Wc[n][t][m] = sum_c W[t][c]*comp[n][c][m]   (padded stride NCP, col 15 = 0)
__global__ __launch_bounds__(256) void wc_kernel(
    const float* __restrict__ whaar, const float* __restrict__ comp, float* __restrict__ Wc) {
  int idx = blockIdx.x * 256 + threadIdx.x;
  if (idx >= Nh * Tc * NCP) return;
  int m = idx & (NCP - 1);
  int t = (idx / NCP) % Tc;
  int n = idx / (NCP * Tc);
  float a = 0.f;
  if (m < NCc)
    for (int c = 0; c < NCc; ++c)
      a += whaar[t * NCP + c] * comp[(size_t)n * NCc * NCc + c * NCc + m];
  Wc[idx] = a;
}

// ------------------------------------------------ fused two-pass attention
__global__ __launch_bounds__(256) void attn_kernel(
    const u16* __restrict__ qb, const u16* __restrict__ kb, const u16* __restrict__ vtb,
    const u16* __restrict__ vnb, const u16* __restrict__ svb,
    const float* __restrict__ Wc, const float* __restrict__ whaar,
    const float* __restrict__ kn, const float* __restrict__ tau,
    u16* __restrict__ ybf) {
  int blk = blockIdx.x;
  int nh = blk >> 6;
  int t0 = (blk & 63) * 16;
  int h = nh % Hc, b = nh / Hc;
  int tid = threadIdx.x, lane = tid & 31, wave = tid >> 5;
  int g = lane >> 4, lm = lane & 15;

  __shared__ __align__(16) float wc_s[16][NCP];
  __shared__ float ssum[16], ssq[16], smean[16], sstd[16], sdenom[16];
  __shared__ float ylds[16][Dc];
  __shared__ __align__(16) u16 plds[8][16][40];  // stride 40 halves = 80B (16B aligned)

  for (int i = tid; i < 16 * NCP; i += 256)
    wc_s[i / NCP][i % NCP] = Wc[((size_t)nh * Tc + t0 + i / NCP) * NCP + i % NCP];
  if (tid < 16) { ssum[tid] = 0.f; ssq[tid] = 0.f; sdenom[tid] = 0.f; }
  for (int i = tid; i < 16 * Dc; i += 256) ylds[i / Dc][i % Dc] = 0.f;
  __syncthreads();

  float tau_h = tau[h];
  const u16* qbase = qb + (size_t)nh * Tc * Dc;
  const u16* kbase = kb + (size_t)nh * Tc * Dc;
  const u16* vnbase = vnb + (size_t)nh * Tc * Dc;
  const u16* svbase = svb + (size_t)nh * Tc * Dc;
  const u16* vtbase = vtb + (size_t)nh * Dc * Tc;

  V16 qa0, qa1, sa0, sa1;   // A-fragments: q rows and Sv rows (K=64 -> 2 frags each)
  {
    const u16* qrow = qbase + (size_t)(t0 + lm) * Dc;
    const u16* srow = svbase + (size_t)(t0 + lm) * Dc;
    load_frag16(qa0, qrow, g);      load_frag16(qa1, qrow + 32, g);
    load_frag16(sa0, srow, g);      load_frag16(sa1, srow + 32, g);
  }

  // Score tile (16x16 at column base ss0) in C-layout regs; identical in both passes.
  auto score_tile = [&](int ss0) -> v8f {
    const v8f vz = {0.f,0.f,0.f,0.f,0.f,0.f,0.f,0.f};
    V16 b0, b1;
    const u16* vr = vnbase + (size_t)(ss0 + lm) * Dc;
    load_frag16(b0, vr, g);
    load_frag16(b1, vr + 32, g);
    v8f w = vz;
    w = wmma_bf16(sa0.v, b0.v, w);
    w = wmma_bf16(sa1.v, b1.v, w);
    v8f qk = vz;
    bool needQK = (ss0 + 15 >= t0 - NEARc);
    if (needQK) {
      V16 kb0, kb1;
      const u16* kr = kbase + (size_t)(ss0 + lm) * Dc;
      load_frag16(kb0, kr, g);
      load_frag16(kb1, kr + 32, g);
      qk = wmma_bf16(qa0.v, kb0.v, qk);
      qk = wmma_bf16(qa1.v, kb1.v, qk);
    }
    int s = ss0 + lm;
    // rank-15 Haar term: Wc_tile(16x16pad) @ Whaar_s^T, on the f32 matrix path
    v8f af = vz;
#if __has_builtin(__builtin_amdgcn_wmma_f32_16x16x4_f32)
#pragma unroll
    for (int k4 = 0; k4 < NCP; k4 += 4) {
      v2f a2, b2;
      a2[0] = wc_s[lm][k4 + 2 * g];
      a2[1] = wc_s[lm][k4 + 2 * g + 1];
      const float* wrow = whaar + (size_t)s * NCP + k4 + 2 * g;
      b2[0] = wrow[0]; b2[1] = wrow[1];
      af = __builtin_amdgcn_wmma_f32_16x16x4_f32(false, a2, false, b2, (short)0, af, false, false);
    }
#else
    {
      float wh[NCc];
#pragma unroll
      for (int c = 0; c < NCc; ++c) wh[c] = whaar[(size_t)s * NCP + c];
#pragma unroll
      for (int r = 0; r < 8; ++r) {
        float acc = 0.f;
#pragma unroll
        for (int c = 0; c < NCc; ++c) acc += wc_s[g * 8 + r][c] * wh[c];
        af[r] = acc;
      }
    }
#endif
    v8f out;
#pragma unroll
    for (int r = 0; r < 8; ++r) {
      int t = t0 + g * 8 + r;
      float dist = fabsf((float)(t - s));
      float geo = 0.5f * w[r] * __expf(-dist * 0.01f / tau_h);
      out[r] = (dist <= (float)NEARc) ? (qk[r] + geo) : (af[r] + geo);
    }
    return out;
  };

  int nchunks = (t0 + 16 + 31) >> 5;   // only s-chunks with s0 <= t0+15 matter

  // ---- PASS 1: per-row sum / sum-of-squares over finite (causal) entries
  float psum[8], psq[8];
#pragma unroll
  for (int r = 0; r < 8; ++r) { psum[r] = 0.f; psq[r] = 0.f; }
  for (int cj = wave; cj < nchunks; cj += 8) {
    int s0 = cj * 32;
    for (int hv = 0; hv < 2; ++hv) {
      int ss0 = s0 + 16 * hv;
      if (ss0 > t0 + 15) continue;       // tile fully above diagonal
      v8f sc = score_tile(ss0);
      int s = ss0 + lm;
#pragma unroll
      for (int r = 0; r < 8; ++r) {
        int t = t0 + g * 8 + r;
        if (s <= t) { float v = sc[r]; psum[r] += v; psq[r] += v * v; }
      }
    }
  }
#pragma unroll
  for (int r = 0; r < 8; ++r) {
    atomicAdd(&ssum[g * 8 + r], psum[r]);
    atomicAdd(&ssq[g * 8 + r], psq[r]);
  }
  __syncthreads();

  if (tid < 16) {
    int t = t0 + tid;
    const u16* qrow = qbase + (size_t)t * Dc;
    const float* knh = kn + h * Dc;
    float ns = 0.f;
    for (int d = 0; d < Dc; ++d) ns += bf2f(qrow[d]) * knh[d];  // null score
    float n = (float)(t + 2);                                    // t+1 causal + 1 null
    float sum = ssum[tid] + ns, sq = ssq[tid] + ns * ns;
    float mean = sum / n;
    float var = sq / n - mean * mean;
    float stdv = sqrtf(fmaxf(var, 1e-6f));
    smean[tid] = mean; sstd[tid] = stdv;
    sdenom[tid] = __expf((ns - mean) / stdv);  // null contributes to denom only
  }
  __syncthreads();

  // ---- PASS 2: recompute scores, exponentiate, probs@V via WMMA
  const v8f vz = {0.f,0.f,0.f,0.f,0.f,0.f,0.f,0.f};
  v8f yacc[4];
#pragma unroll
  for (int dt = 0; dt < 4; ++dt) yacc[dt] = vz;
  float pden[8];
#pragma unroll
  for (int r = 0; r < 8; ++r) pden[r] = 0.f;
  for (int cj = wave; cj < nchunks; cj += 8) {
    int s0 = cj * 32;
    for (int hv = 0; hv < 2; ++hv) {
      int ss0 = s0 + 16 * hv;
      int s = ss0 + lm;
      if (ss0 > t0 + 15) {
#pragma unroll
        for (int r = 0; r < 8; ++r) plds[wave][g * 8 + r][16 * hv + lm] = 0;
        continue;
      }
      v8f sc = score_tile(ss0);
#pragma unroll
      for (int r = 0; r < 8; ++r) {
        int m = g * 8 + r;
        int t = t0 + m;
        float p = 0.f;
        if (s <= t) { p = __expf((sc[r] - smean[m]) / sstd[m]); pden[r] += p; }
        plds[wave][m][16 * hv + lm] = f2bf(p);
      }
    }
    // probs (16x32) A-frag from per-wave LDS tile; V (32x64) B-frags from V^T rows
    V16 pa;
    load_frag16(pa, &plds[wave][lm][0], g);
#pragma unroll
    for (int dt = 0; dt < 4; ++dt) {
      V16 bv;
      const u16* vrow = vtbase + (size_t)(dt * 16 + lm) * Tc + s0;
      load_frag16(bv, vrow, g);
      yacc[dt] = wmma_bf16(pa.v, bv.v, yacc[dt]);
    }
  }
#pragma unroll
  for (int r = 0; r < 8; ++r) atomicAdd(&sdenom[g * 8 + r], pden[r]);
#pragma unroll
  for (int dt = 0; dt < 4; ++dt)
#pragma unroll
    for (int r = 0; r < 8; ++r)
      atomicAdd(&ylds[g * 8 + r][dt * 16 + lm], yacc[dt][r]);
  __syncthreads();

  for (int i = tid; i < 16 * Dc; i += 256) {
    int m = i / Dc, d = i % Dc;
    float val = ylds[m][d] / sdenom[m];
    ybf[((size_t)(b * Tc + t0 + m)) * Cc + h * Dc + d] = f2bf(val);
  }
}

// ================================================================= launcher
extern "C" void kernel_launch(void* const* d_in, const int* in_sizes, int n_in,
                              void* d_out, int out_size, void* d_ws, size_t ws_size,
                              hipStream_t stream) {
  (void)in_sizes; (void)n_in; (void)out_size; (void)ws_size;
  const float* x       = (const float*)d_in[0];
  const float* Wq      = (const float*)d_in[1];
  const float* Wk      = (const float*)d_in[2];
  const float* Wv      = (const float*)d_in[3];
  const float* Wo      = (const float*)d_in[4];
  const float* ln_w    = (const float*)d_in[5];
  const float* ln_b    = (const float*)d_in[6];
  const float* k_null  = (const float*)d_in[7];
  const float* Amat    = (const float*)d_in[8];
  const float* log_tau = (const float*)d_in[9];
  float* out = (float*)d_out;

  char* ws = (char*)d_ws;
  size_t off = 0;
  auto alloc = [&](size_t bytes) -> char* {
    char* p = ws + off;
    off += (bytes + 255) & ~(size_t)255;
    return p;
  };
  float* whaar = (float*)alloc((size_t)Tc * NCP * 4);
  float* cosT  = (float*)alloc((size_t)Tc * HALFc * 4);
  float* sinT  = (float*)alloc((size_t)Tc * HALFc * 4);
  float* kn    = (float*)alloc((size_t)Hc * Dc * 4);
  float* tau   = (float*)alloc((size_t)Hc * 4);
  u16* x_bf    = (u16*)alloc((size_t)BTc * Cc * 2);
  u16* xln_bf  = (u16*)alloc((size_t)BTc * Cc * 2);
  u16* Wqb     = (u16*)alloc((size_t)Cc * Cc * 2);
  u16* Wkb     = (u16*)alloc((size_t)Cc * Cc * 2);
  u16* Wvb     = (u16*)alloc((size_t)Cc * Cc * 2);
  u16* Wob     = (u16*)alloc((size_t)Cc * Cc * 2);
  float* Qf    = (float*)alloc((size_t)BTc * Cc * 4);
  float* Kf    = (float*)alloc((size_t)BTc * Cc * 4);
  float* Vf    = (float*)alloc((size_t)BTc * Cc * 4);
  u16* qb      = (u16*)alloc((size_t)Nh * Tc * Dc * 2);
  u16* kb      = (u16*)alloc((size_t)Nh * Tc * Dc * 2);
  u16* vtb     = (u16*)alloc((size_t)Nh * Dc * Tc * 2);
  u16* vnb     = (u16*)alloc((size_t)Nh * Tc * Dc * 2);
  u16* svb     = (u16*)alloc((size_t)Nh * Tc * Dc * 2);
  float* comp  = (float*)alloc((size_t)Nh * NCc * NCc * 4);
  float* Wc    = (float*)alloc((size_t)Nh * Tc * NCP * 4);
  u16* ybf     = (u16*)alloc((size_t)BTc * Cc * 2);

  int tot0 = Tc * NCP + Tc * HALFc + Hc;
  build_tables<<<(tot0 + 255) / 256, 256, 0, stream>>>(whaar, cosT, sinT, k_null, log_tau, kn, tau);
  pack_x_ln<<<BTc, 256, 0, stream>>>(x, ln_w, ln_b, x_bf, xln_bf);
  pack_wt<<<(4 * Cc * Cc) / 256, 256, 0, stream>>>(Wq, Wk, Wv, Wo, Wqb, Wkb, Wvb, Wob);
  dim3 ggrid(Cc / 64, BTc / 64);
  gemm_bf16_wmma<<<ggrid, 128, 0, stream>>>(x_bf,   Wqb, Qf, BTc, Cc, Cc);
  gemm_bf16_wmma<<<ggrid, 128, 0, stream>>>(x_bf,   Wkb, Kf, BTc, Cc, Cc);
  gemm_bf16_wmma<<<ggrid, 128, 0, stream>>>(xln_bf, Wvb, Vf, BTc, Cc, Cc);
  post_qkv<<<Nh * Tc, 64, 0, stream>>>(Qf, Kf, Vf, x, cosT, sinT, Amat, qb, kb, vtb, vnb, svb);
  haar_qkfar<<<Nh, 256, 0, stream>>>(qb, kb, whaar, comp);
  wc_kernel<<<(Nh * Tc * NCP + 255) / 256, 256, 0, stream>>>(whaar, comp, Wc);
  attn_kernel<<<Nh * (Tc / 16), 256, 0, stream>>>(qb, kb, vtb, vnb, svb, Wc, whaar, kn, tau, ybf);
  gemm_bf16_wmma<<<ggrid, 128, 0, stream>>>(ybf, Wob, out, BTc, Cc, Cc);
}
